// VanillaLSTM_72155450573208
// MI455X (gfx1250) — matrix operand order repaired and checked by reading, not compiled, not run on hardware
//
#include <hip/hip_runtime.h>

// ---------------------------------------------------------------------------
// VanillaLSTM on MI455X (gfx1250), wave32 + WMMA bf16 + TDM staging.
//   B=8192, T=64, D_IN=2, E=128, H=128, D_OUT=5
// Gate GEMM per step: [e|h](8192x256) @ Wcat^T(256x512) via
// v_wmma_f32_16x16x32_bf16 (B-fragments read from global Wcat, A from LDS).
// h_prev tile staged into LDS by the Tensor Data Mover with LDS padding to
// interleave into the [e|h] A-stage layout. All BN folded analytically.
// ---------------------------------------------------------------------------

#define BB     8192
#define TT     64
#define EE     128
#define HH     128
#define NG     512     // 4*H gates
#define KK2    256     // E + H
#define DOUT   5
#define EPSBN  1e-5f

typedef __attribute__((ext_vector_type(16))) __bf16 bf16x16;
typedef __attribute__((ext_vector_type(8)))  float  f32x8;
typedef __attribute__((ext_vector_type(4)))  unsigned int u32x4;
typedef __attribute__((ext_vector_type(8)))  int          i32x8;
typedef __attribute__((ext_vector_type(4)))  int          i32x4;

struct Frag32 { uint4 lo, hi; };   // 32 bytes = 16 bf16 = one WMMA operand/lane

static __device__ __forceinline__ unsigned short f2bf(float f) {
  unsigned u = __float_as_uint(f);
  unsigned r = ((u >> 16) & 1u) + 0x7FFFu;   // round-to-nearest-even
  return (unsigned short)((u + r) >> 16);
}

#if defined(__HIP_DEVICE_COMPILE__) && __has_builtin(__builtin_amdgcn_tensor_load_to_lds)
#define USE_TDM 1
#else
#define USE_TDM 0
#endif

// ---------------------------------------------------------------------------
// 0) zero recurrent state
__global__ void init_state_kernel(unsigned short* hbf, float* cbuf) {
  int i = blockIdx.x * blockDim.x + threadIdx.x;
  if (i < BB * HH) { hbf[i] = 0; cbuf[i] = 0.f; }
}

// 1) Wcat bf16 (512 x 256, n-major / k-contiguous): [W_ih | W_hh]
__global__ void wprep_kernel(const float* __restrict__ W_ih,
                             const float* __restrict__ W_hh,
                             unsigned short* __restrict__ Wcat) {
  int i = blockIdx.x * blockDim.x + threadIdx.x;
  if (i >= NG * KK2) return;
  int n = i >> 8, k = i & 255;
  float v = (k < HH) ? W_ih[n * HH + k] : W_hh[n * HH + (k - HH)];
  Wcat[i] = f2bf(v);
}

// 2) per-step raw moments of the 2 input channels (deterministic tree)
__global__ __launch_bounds__(256)
void moments_kernel(const float* __restrict__ inputs, float* __restrict__ moms) {
  __shared__ float red[256];
  const int t = blockIdx.x, tid = threadIdx.x;
  float s0 = 0, s1 = 0, s00 = 0, s11 = 0, s01 = 0;
  for (int b = tid; b < BB; b += 256) {
    float x0 = inputs[(b * TT + t) * 2 + 0];
    float x1 = inputs[(b * TT + t) * 2 + 1];
    s0 += x0; s1 += x1; s00 += x0 * x0; s11 += x1 * x1; s01 += x0 * x1;
  }
  float vals[5] = { s0, s1, s00, s11, s01 };
  for (int c = 0; c < 5; ++c) {
    red[tid] = vals[c]; __syncthreads();
    for (int s = 128; s > 0; s >>= 1) {
      if (tid < s) red[tid] += red[tid + s];
      __syncthreads();
    }
    if (tid == 0) moms[t * 5 + c] = red[0];
    __syncthreads();
  }
}

// 3) fold inputBN + embed Linear + per-step embedBN into e = relu(p0*x0+p1*x1+p2)
__global__ void embed_params_kernel(const float* __restrict__ moms,
                                    const float* __restrict__ in_gamma,
                                    const float* __restrict__ in_beta,
                                    const float* __restrict__ W_emb,
                                    const float* __restrict__ b_emb,
                                    const float* __restrict__ emb_gamma,
                                    const float* __restrict__ emb_beta,
                                    float* __restrict__ p0,
                                    float* __restrict__ p1,
                                    float* __restrict__ p2) {
  int gid = blockIdx.x * blockDim.x + threadIdx.x;
  if (gid >= TT * EE) return;
  int t = gid >> 7, j = gid & 127;

  float S0 = 0, S1 = 0, S00 = 0, S11 = 0, S01 = 0;
  for (int tt = 0; tt < TT; ++tt) {
    S0 += moms[tt * 5 + 0]; S1 += moms[tt * 5 + 1];
    S00 += moms[tt * 5 + 2]; S11 += moms[tt * 5 + 3]; S01 += moms[tt * 5 + 4];
  }
  const float NT = (float)BB * (float)TT;
  float m0 = S0 / NT, m1 = S1 / NT;
  float v0 = S00 / NT - m0 * m0, v1 = S11 / NT - m1 * m1;
  float q0 = rsqrtf(v0 + EPSBN) * in_gamma[0];
  float q1 = rsqrtf(v1 + EPSBN) * in_gamma[1];
  float r0 = in_beta[0] - m0 * q0;
  float r1 = in_beta[1] - m1 * q1;

  const float NB = (float)BB;
  float t0  = moms[t * 5 + 0] / NB, t1 = moms[t * 5 + 1] / NB;
  float t00 = moms[t * 5 + 2] / NB - t0 * t0;
  float t11 = moms[t * 5 + 3] / NB - t1 * t1;
  float t01 = moms[t * 5 + 4] / NB - t0 * t1;

  float mn0 = t0 * q0 + r0, mn1 = t1 * q1 + r1;
  float vn00 = q0 * q0 * t00, vn11 = q1 * q1 * t11, vn01 = q0 * q1 * t01;

  float W0 = W_emb[j * 2 + 0], W1 = W_emb[j * 2 + 1];
  float mean_y = W0 * mn0 + W1 * mn1 + b_emb[j];
  float var_y  = W0 * W0 * vn00 + 2.f * W0 * W1 * vn01 + W1 * W1 * vn11;
  float kk = emb_gamma[j] * rsqrtf(var_y + EPSBN);

  p0[gid] = kk * W0 * q0;
  p1[gid] = kk * W1 * q1;
  p2[gid] = kk * (W0 * r0 + W1 * r1 + b_emb[j] - mean_y) + emb_beta[j];
}

// ---------------------------------------------------------------------------
// 4) one fused LSTM step: e-gen + TDM h-stage -> WMMA gate GEMM -> pointwise
//    -> out Linear. grid 128 blocks x 256 threads; block = 4 M-tiles of 16.
__global__ __launch_bounds__(256)
void lstm_step_kernel(const float* __restrict__ inputs,
                      const float* __restrict__ p0,
                      const float* __restrict__ p1,
                      const float* __restrict__ p2,
                      const unsigned short* __restrict__ Wcat,
                      unsigned short* __restrict__ hbf,
                      float* __restrict__ cbuf,
                      const float* __restrict__ b_ih,
                      const float* __restrict__ b_hh,
                      const float* __restrict__ W_out,
                      const float* __restrict__ b_out,
                      float* __restrict__ out,
                      float* __restrict__ hout,
                      float* __restrict__ cout,
                      int t) {
  __shared__ __align__(16) unsigned short Astage[16 * KK2];  // 8 KB, reused as f32 h
  __shared__ __align__(16) float gates[16 * NG];             // 32 KB

  const int tid  = threadIdx.x;
  const int lane = tid & 31;
  const int wv   = tid >> 5;        // 8 waves, wave wv owns N-tiles [4w,4w+4)

  const int rowbase0 = blockIdx.x * 64;

  for (int mt = 0; mt < 4; ++mt) {
    const int rowbase = rowbase0 + mt * 16;

#if USE_TDM
    // ---- TDM: DMA h_prev tile (16 rows x 128 bf16) into the h-half of the
    //      A-stage. LDS padding (64 DW data + 64 DW pad) interleaves the
    //      256 B h rows into the 512 B-stride [e|h] layout. Wave 0 issues.
    if (wv == 0) {
      unsigned long long ga =
          (unsigned long long)(uintptr_t)(hbf + (size_t)rowbase * HH);
      unsigned lds_base = (unsigned)(uintptr_t)(&Astage[EE]);  // h-half, row 0
      u32x4 g0;
      g0.x = 1u;                                    // count=1, user descriptor
      g0.y = lds_base;                              // lds_addr (bytes)
      g0.z = (unsigned)(ga & 0xFFFFFFFFull);        // global_addr[31:0]
      g0.w = (unsigned)((ga >> 32) & 0x01FFFFFFull) // global_addr[56:32]
             | (2u << 30);                          // type = 2 (image)
      i32x8 g1;
      g1[0] = (int)((1u << 16)      // data_size = 2 bytes
                  | (1u << 20)      // pad_enable
                  | (5u << 22)      // pad_interval: 64 DWORDs of data
                  | (63u << 25));   // pad_amount: 64 DWORDs of pad
      g1[1] = (int)(HH << 16);      // tensor_dim0 = 128 (elements)
      g1[2] = (int)((BB & 0xFFFF) << 16);  // tensor_dim1 = 8192 (low 16)
      g1[3] = (int)((BB >> 16) | (HH << 16));  // dim1 hi | tile_dim0 = 128
      g1[4] = 16;                   // tile_dim1 = 16 rows (tile_dim2 = 0)
      g1[5] = HH;                   // tensor_dim0_stride = 128 (low 32)
      g1[6] = 0;                    // stride hi | dim1_stride lo (unused, 2D)
      g1[7] = 0;
      i32x4 gz4 = { 0, 0, 0, 0 };   // groups 2/3 unused for 2D tile
      i32x8 gz8 = { 0, 0, 0, 0, 0, 0, 0, 0 };
      __builtin_amdgcn_tensor_load_to_lds(g0, g1, gz4, gz4, gz8, 0);
    }
#endif

    // ---- stage e_t (bf16, packed pairs) into the e-half of A ----
    for (int idx = tid; idx < 16 * (EE / 2); idx += 256) {
      int r = idx >> 6, j = (idx & 63) * 2;
      int b = rowbase + r;
      float x0 = inputs[(b * TT + t) * 2 + 0];
      float x1 = inputs[(b * TT + t) * 2 + 1];
      float ea = fmaxf(p0[t * EE + j]     * x0 + p1[t * EE + j]     * x1 + p2[t * EE + j],     0.f);
      float eb = fmaxf(p0[t * EE + j + 1] * x0 + p1[t * EE + j + 1] * x1 + p2[t * EE + j + 1], 0.f);
      unsigned packed = (unsigned)f2bf(ea) | ((unsigned)f2bf(eb) << 16);
      *(unsigned*)(&Astage[r * KK2 + j]) = packed;
    }

#if USE_TDM
    if (wv == 0) __builtin_amdgcn_s_wait_tensorcnt(0);
#else
    for (int idx = tid; idx < 16 * HH; idx += 256) {
      int r = idx >> 7, j = idx & 127;
      Astage[r * KK2 + EE + j] = hbf[(rowbase + r) * HH + j];
    }
#endif
    __syncthreads();

    // ---- gate GEMM: 16x512 += (16x256) x (256x512) via WMMA bf16 ----
    {
      f32x8 acc[4] = {};
      const int arow  = lane & 15;
      const int ahalf = lane >> 4;
      // preload all 8 A-fragments (batch the DS waits once per M-tile)
      Frag32 fa[8];
#pragma unroll
      for (int kk = 0; kk < 8; ++kk) {
        const unsigned short* ap = Astage + arow * KK2 + kk * 32 + ahalf * 8;
        fa[kk].lo = *(const uint4*)(ap);        // K k..k+7
        fa[kk].hi = *(const uint4*)(ap + 16);   // K k+16..k+23
      }
#pragma unroll
      for (int kk = 0; kk < 8; ++kk) {
        bf16x16 a = __builtin_bit_cast(bf16x16, fa[kk]);
#pragma unroll
        for (int i = 0; i < 4; ++i) {
          int n = (wv * 4 + i) * 16 + (lane & 15);
          const unsigned short* bp = Wcat + n * KK2 + kk * 32 + (lane >> 4) * 8;
          Frag32 fb;
          fb.lo = *(const uint4*)(bp);          // K k..k+7
          fb.hi = *(const uint4*)(bp + 16);     // K k+16..k+23
          bf16x16 bm = __builtin_bit_cast(bf16x16, fb);
          acc[i] = __builtin_amdgcn_wmma_f32_16x16x32_bf16(
              false, a, false, bm, (short)0, acc[i], false, false);
        }
      }
      // D layout: VGPR r -> M = r + 8*(lane>>4), N = lane&15
#pragma unroll
      for (int i = 0; i < 4; ++i) {
        int nb = (wv * 4 + i) * 16 + (lane & 15);
        int mb = (lane >> 4) * 8;
#pragma unroll
        for (int r = 0; r < 8; ++r) gates[(mb + r) * NG + nb] = acc[i][r];
      }
    }
    __syncthreads();

    // ---- LSTM pointwise (c in f32 ws, h to bf16 ws + f32 LDS) ----
    float* hlds = (float*)Astage;   // reuse: 16*128 f32 = 8 KB (A consumed)
    for (int idx = tid; idx < 16 * HH; idx += 256) {
      int r = idx >> 7, k = idx & 127;
      int b = rowbase + r;
      float gi = gates[r * NG + k]        + b_ih[k]        + b_hh[k];
      float gf = gates[r * NG + 128 + k]  + b_ih[128 + k]  + b_hh[128 + k];
      float gg = gates[r * NG + 256 + k]  + b_ih[256 + k]  + b_hh[256 + k];
      float go = gates[r * NG + 384 + k]  + b_ih[384 + k]  + b_hh[384 + k];
      float si = 1.f / (1.f + __expf(-gi));
      float sf = 1.f / (1.f + __expf(-gf));
      float so = 1.f / (1.f + __expf(-go));
      float tg = tanhf(gg);
      float cold = cbuf[b * HH + k];
      float cn = sf * cold + si * tg;
      float hn = so * tanhf(cn);
      cbuf[b * HH + k] = cn;
      hbf[b * HH + k]  = f2bf(hn);
      hlds[r * HH + k] = hn;
      if (t == TT - 1) { hout[b * HH + k] = hn; cout[b * HH + k] = cn; }
    }
    __syncthreads();

    // ---- out Linear (128->5), raw z; BN applied later globally ----
    if (tid < 16 * DOUT) {
      int r = tid / DOUT, o = tid % DOUT;
      float s = b_out[o];
      for (int k = 0; k < HH; ++k) s += hlds[r * HH + k] * W_out[o * HH + k];
      out[((rowbase + r) * TT + t) * DOUT + o] = s;
    }
    __syncthreads();
  }
}

// ---------------------------------------------------------------------------
// 5) per-(t,channel) output BN stats over B (deterministic tree)
__global__ __launch_bounds__(256)
void outstats_kernel(const float* __restrict__ out, float* __restrict__ stats) {
  __shared__ float red[256];
  const int t = blockIdx.x, tid = threadIdx.x;
  float s[DOUT] = {0, 0, 0, 0, 0}, s2[DOUT] = {0, 0, 0, 0, 0};
  for (int b = tid; b < BB; b += 256) {
    const float* z = out + (b * TT + t) * DOUT;
#pragma unroll
    for (int c = 0; c < DOUT; ++c) { float v = z[c]; s[c] += v; s2[c] += v * v; }
  }
  for (int c = 0; c < DOUT; ++c) {
    red[tid] = s[c]; __syncthreads();
    for (int k = 128; k > 0; k >>= 1) { if (tid < k) red[tid] += red[tid + k]; __syncthreads(); }
    if (tid == 0) stats[t * 10 + c] = red[0];
    __syncthreads();
    red[tid] = s2[c]; __syncthreads();
    for (int k = 128; k > 0; k >>= 1) { if (tid < k) red[tid] += red[tid + k]; __syncthreads(); }
    if (tid == 0) stats[t * 10 + 5 + c] = red[0];
    __syncthreads();
  }
}

// 6) in-place output BN normalize
__global__ void outnorm_kernel(float* __restrict__ out,
                               const float* __restrict__ stats,
                               const float* __restrict__ og,
                               const float* __restrict__ ob) {
  int idx = blockIdx.x * blockDim.x + threadIdx.x;
  if (idx >= BB * TT * DOUT) return;
  int o = idx % DOUT;
  int t = (idx / DOUT) & (TT - 1);
  float mean = stats[t * 10 + o] / (float)BB;
  float var  = stats[t * 10 + 5 + o] / (float)BB - mean * mean;
  out[idx] = (out[idx] - mean) * rsqrtf(var + EPSBN) * og[o] + ob[o];
}

// ---------------------------------------------------------------------------
extern "C" void kernel_launch(void* const* d_in, const int* in_sizes, int n_in,
                              void* d_out, int out_size, void* d_ws, size_t ws_size,
                              hipStream_t stream) {
  const float* inputs    = (const float*)d_in[0];
  const float* in_gamma  = (const float*)d_in[1];
  const float* in_beta   = (const float*)d_in[2];
  const float* W_emb     = (const float*)d_in[3];
  const float* b_emb     = (const float*)d_in[4];
  const float* emb_gamma = (const float*)d_in[5];
  const float* emb_beta  = (const float*)d_in[6];
  const float* W_ih      = (const float*)d_in[7];
  const float* W_hh      = (const float*)d_in[8];
  const float* b_ih      = (const float*)d_in[9];
  const float* b_hh      = (const float*)d_in[10];
  const float* W_out     = (const float*)d_in[11];
  const float* b_out     = (const float*)d_in[12];
  const float* out_gamma = (const float*)d_in[13];
  const float* out_beta  = (const float*)d_in[14];

  float* out  = (float*)d_out;
  float* hout = out + (size_t)BB * TT * DOUT;
  float* cout = hout + (size_t)BB * HH;

  // workspace layout (~6.7 MB)
  char* ws = (char*)d_ws;
  size_t off = 0;
  unsigned short* Wcat = (unsigned short*)(ws + off); off += (size_t)NG * KK2 * 2;   // 262144
  float* p0   = (float*)(ws + off); off += (size_t)TT * EE * 4;                      // 32768
  float* p1   = (float*)(ws + off); off += (size_t)TT * EE * 4;
  float* p2   = (float*)(ws + off); off += (size_t)TT * EE * 4;
  float* moms = (float*)(ws + off); off += (size_t)TT * 5 * 4 + 192; off &= ~255ull;
  unsigned short* hbf = (unsigned short*)(ws + off); off += (size_t)BB * HH * 2;     // 2 MB
  float* cbuf = (float*)(ws + off); off += (size_t)BB * HH * 4;                      // 4 MB
  float* stats = (float*)(ws + off);

  init_state_kernel<<<(BB * HH + 255) / 256, 256, 0, stream>>>(hbf, cbuf);
  wprep_kernel<<<(NG * KK2 + 255) / 256, 256, 0, stream>>>(W_ih, W_hh, Wcat);
  moments_kernel<<<TT, 256, 0, stream>>>(inputs, moms);
  embed_params_kernel<<<(TT * EE + 255) / 256, 256, 0, stream>>>(
      moms, in_gamma, in_beta, W_emb, b_emb, emb_gamma, emb_beta, p0, p1, p2);

  for (int t = 0; t < TT; ++t) {
    lstm_step_kernel<<<BB / 64, 256, 0, stream>>>(
        inputs, p0, p1, p2, Wcat, hbf, cbuf, b_ih, b_hh, W_out, b_out,
        out, hout, cout, t);
  }

  outstats_kernel<<<TT, 256, 0, stream>>>(out, stats);
  outnorm_kernel<<<(BB * TT * DOUT + 255) / 256, 256, 0, stream>>>(
      out, stats, out_gamma, out_beta);
}
